// Architecture_67181878444994
// MI455X (gfx1250) — matrix operand order repaired and checked by reading, not compiled, not run on hardware
//
#include <hip/hip_runtime.h>
#include <hip/hip_bf16.h>

// ---------------------------------------------------------------------------
// STU forward (2 layers) for MI455X / gfx1250, wave32, WMMA bf16.
// Spectral conv: causal-Toeplitz WMMA GEMM with PRECOMPUTED per-diagonal
// packed A fragments (block-Toeplitz dedup, L2-resident table) fused with the
// m_phi contraction and AR taps. Batched WMMA scan. Async global->LDS staging
// for weight tiles.
// ---------------------------------------------------------------------------

#define B_    8
#define T_    2048
#define D_    128
#define KE    24
#define KU    3
#define LL    2
#define DT    64
#define ROWS  (B_ * T_)   // 16384
#define NDG   64          // distinct 32-aligned diagonals (-32 .. 1984)

typedef __attribute__((ext_vector_type(16))) __bf16 v16bf;
typedef __attribute__((ext_vector_type(8)))  float  v8f;
typedef __attribute__((ext_vector_type(4)))  int    v4i;

union AFrag { v16bf v; __bf16 e[16]; unsigned u[8]; uint4 q[2]; };

#if defined(__has_builtin)
#if __has_builtin(__builtin_amdgcn_global_load_async_to_lds_b128) && \
    __has_builtin(__builtin_amdgcn_s_wait_asynccnt)
#define USE_ASYNC 1
#endif
#endif

__device__ inline void stage_b128(const void* gsrc, void* ldst) {
#ifdef USE_ASYNC
  __builtin_amdgcn_global_load_async_to_lds_b128(
      (__attribute__((address_space(1))) v4i*)gsrc,
      (__attribute__((address_space(3))) v4i*)ldst, 0, 0);
#else
  *(uint4*)ldst = *(const uint4*)gsrc;
#endif
}
__device__ inline void stage_wait() {
#ifdef USE_ASYNC
  __builtin_amdgcn_s_wait_asynccnt(0);
#endif
}

__device__ inline v8f vzero() {
  v8f z;
#pragma unroll
  for (int i = 0; i < 8; ++i) z[i] = 0.0f;
  return z;
}

__device__ inline v8f wmma_bf16(v16bf a, v16bf b, v8f c) {
  return __builtin_amdgcn_wmma_f32_16x16x32_bf16(false, a, false, b, (short)0, c,
                                                 false, false);
}

// A fragment: 16x32 bf16 tile, row-major [M][K], leading dim ld (mult of 8).
__device__ inline v16bf load_a_frag(const __bf16* base, int ld) {
  int lane = threadIdx.x & 31;
  const __bf16* rp = base + (lane & 15) * ld + (lane >> 4) * 8;
  AFrag f;
  f.q[0] = *(const uint4*)(rp);
  f.q[1] = *(const uint4*)(rp + 16);
  return f.v;
}

// B fragment from TRANSPOSED storage [N][K] (ldk mult of 8).
__device__ inline v16bf load_bt_frag(const __bf16* baseT, int ldk) {
  int lane = threadIdx.x & 31;
  const __bf16* rp = baseT + (lane & 15) * ldk + (lane >> 4) * 8;
  AFrag f;
  f.q[0] = *(const uint4*)(rp);
  f.q[1] = *(const uint4*)(rp + 16);
  return f.v;
}

// ------------------------------ prep kernels -------------------------------

__global__ void k_cvt(const float* __restrict__ s, __bf16* __restrict__ d, int n) {
  for (int i = blockIdx.x * blockDim.x + threadIdx.x; i < n;
       i += gridDim.x * blockDim.x)
    d[i] = (__bf16)s[i];
}

// embw_bt[n][k] = emb_w[k][n]
__global__ void k_prep_embw(const float* __restrict__ s, __bf16* __restrict__ d) {
  for (int idx = blockIdx.x * blockDim.x + threadIdx.x; idx < D_ * D_;
       idx += gridDim.x * blockDim.x) {
    int k = idx & 127, n = idx >> 7;
    d[idx] = (__bf16)s[(size_t)k * D_ + n];
  }
}

// projw_bt[n][k] = proj_w[k][n]
__global__ void k_prep_proj(const float* __restrict__ s, __bf16* __restrict__ d) {
  for (int idx = blockIdx.x * blockDim.x + threadIdx.x; idx < D_ * DT;
       idx += gridDim.x * blockDim.x) {
    int k = idx & 127, n = idx >> 7;
    d[idx] = (__bf16)s[(size_t)k * DT + n];
  }
}

// w1_bt[l][n(256)][k(128)] = w1[l][k][n]
__global__ void k_prep_w1(const float* __restrict__ s, __bf16* __restrict__ d) {
  int n = LL * D_ * 2 * D_;
  for (int idx = blockIdx.x * blockDim.x + threadIdx.x; idx < n;
       idx += gridDim.x * blockDim.x) {
    int k = idx & 127;
    int c = (idx >> 7) & 255;
    int l = idx >> 15;
    d[idx] = (__bf16)s[((size_t)(l * D_ + k)) * 2 * D_ + c];
  }
}

// mphi_bt[l][kk][o][d] = m_phi[l][kk*128+d][o]
__global__ void k_prep_mphi(const float* __restrict__ s, __bf16* __restrict__ d) {
  int n = LL * KE * D_ * D_;
  for (int idx = blockIdx.x * blockDim.x + threadIdx.x; idx < n;
       idx += gridDim.x * blockDim.x) {
    int dd = idx & 127;
    int o  = (idx >> 7) & 127;
    int kk = (idx >> 14) % KE;
    int l  = idx / (KE << 14);
    d[idx] = (__bf16)s[((size_t)(l * KE + kk) * D_ + dd) * D_ + o];
  }
}

// mu_bt[l][tap][o][i] = m_u[l][o][i][tap]
__global__ void k_prep_mu(const float* __restrict__ s, __bf16* __restrict__ d) {
  int n = LL * KU * D_ * D_;
  for (int idx = blockIdx.x * blockDim.x + threadIdx.x; idx < n;
       idx += gridDim.x * blockDim.x) {
    int i   = idx & 127;
    int o   = (idx >> 7) & 127;
    int tap = (idx >> 14) % KU;
    int l   = idx / (KU << 14);
    d[idx] = (__bf16)s[(((size_t)(l * D_ + o)) * D_ + i) * KU + tap];
  }
}

// eig_bf[k][s] = eig_vecs[s][k] * eig_vals[k]^0.25
__global__ void k_prep_eig(const float* __restrict__ evals,
                           const float* __restrict__ evecs,
                           __bf16* __restrict__ dst) {
  int n = KE * T_;
  for (int idx = blockIdx.x * blockDim.x + threadIdx.x; idx < n;
       idx += gridDim.x * blockDim.x) {
    int s = idx & (T_ - 1);
    int k = idx >> 11;
    dst[idx] = (__bf16)(evecs[(size_t)s * KE + k] * powf(evals[k], 0.25f));
  }
}

// Packed Toeplitz A-fragment table: one entry per (kk, diag, mt, lane, j).
// A-tile for (t0,s0) depends only on diag = t0-s0; dedup across workgroups.
// atab[gid] packs e[2j] = eig[ts], e[2j+1] = eig[ts-1] (zeros below 0).
__global__ void k_prep_afrag(const __bf16* __restrict__ eigb,
                             unsigned* __restrict__ atab) {
  int total = KE * NDG * 4 * 32 * 8;
  for (int gid = blockIdx.x * blockDim.x + threadIdx.x; gid < total;
       gid += gridDim.x * blockDim.x) {
    int j    = gid & 7;
    int lane = (gid >> 3) & 31;
    int mt   = (gid >> 8) & 3;
    int dg   = (gid >> 10) & 63;
    int kk   = gid >> 16;
    int m  = lane & 15;
    int kb = (lane >> 4) * 8;
    int k0 = kb + ((j < 4) ? 0 : 16) + ((j & 3) * 2);
    int ts = (dg - 1) * 32 + mt * 16 + m - k0;   // = t - s
    const __bf16* col = eigb + (size_t)kk * T_;
    union { __bf16 e[2]; unsigned u; } pk;
    pk.e[0] = (ts >= 0) ? col[ts] : (__bf16)0.0f;
    pk.e[1] = (ts >= 1) ? col[ts - 1] : (__bf16)0.0f;
    atab[gid] = pk.u;
  }
}

// ------------------------------ layer norm ---------------------------------

__global__ void __launch_bounds__(256)
k_ln(const float* __restrict__ h, const float* __restrict__ g,
     const float* __restrict__ b, __bf16* __restrict__ hn) {
  int row  = blockIdx.x * 8 + (threadIdx.x >> 5);
  int lane = threadIdx.x & 31;
  float4 v = *(const float4*)(h + (size_t)row * D_ + lane * 4);
  float s = v.x + v.y + v.z + v.w;
#pragma unroll
  for (int o = 16; o > 0; o >>= 1) s += __shfl_xor(s, o, 32);
  float mu = s * (1.0f / 128.0f);
  float dx = v.x - mu, dy = v.y - mu, dz = v.z - mu, dw = v.w - mu;
  float q = dx * dx + dy * dy + dz * dz + dw * dw;
#pragma unroll
  for (int o = 16; o > 0; o >>= 1) q += __shfl_xor(q, o, 32);
  float inv = rsqrtf(q * (1.0f / 128.0f) + 1e-5f);
  int c = lane * 4;
  union { __bf16 e[4]; uint2 u; } o4;
  o4.e[0] = (__bf16)(dx * inv * g[c + 0] + b[c + 0]);
  o4.e[1] = (__bf16)(dy * inv * g[c + 1] + b[c + 1]);
  o4.e[2] = (__bf16)(dz * inv * g[c + 2] + b[c + 2]);
  o4.e[3] = (__bf16)(dw * inv * g[c + 3] + b[c + 3]);
  *(uint2*)(hn + (size_t)row * D_ + c) = o4.u;
}

// --------------------- hn transpose: [b][t][d] -> [b][d][t] ----------------

__global__ void __launch_bounds__(128)
k_tr(const __bf16* __restrict__ hn, __bf16* __restrict__ hnT) {
  __shared__ __attribute__((aligned(16))) __bf16 tile[64][D_ + 8];
  int b = blockIdx.y, t0 = blockIdx.x * 64;
  const __bf16* src = hn + (size_t)b * T_ * D_ + (size_t)t0 * D_;
  __bf16* dst = hnT + (size_t)b * T_ * D_;
  int tid = threadIdx.x;
  for (int i = tid; i < 64 * 16; i += 128) {
    int r = i >> 4, c = i & 15;
    *(uint4*)&tile[r][c * 8] = ((const uint4*)(src + (size_t)r * D_))[c];
  }
  __syncthreads();
  for (int i = tid; i < 128 * 8; i += 128) {
    int d = i >> 3, c = i & 7;
    union { __bf16 e[8]; uint4 q; } o;
#pragma unroll
    for (int j = 0; j < 8; ++j) o.e[j] = tile[c * 8 + j][d];
    ((uint4*)(dst + (size_t)d * T_ + t0))[c] = o.q;
  }
}

// ------------------------------ embedding GEMM -----------------------------

__global__ void __launch_bounds__(128)
k_emb(const float* __restrict__ x, const __bf16* __restrict__ wbt,
      const float* __restrict__ bias, float* __restrict__ h) {
  __shared__ __attribute__((aligned(16))) __bf16 sA[64][D_];   // 16 KB
  __shared__ __attribute__((aligned(16))) __bf16 sB[D_][D_];   // 32 KB [N][K]
  int t0 = blockIdx.x * 64;
  int tid = threadIdx.x, wave = tid >> 5, lane = tid & 31;
  for (int i = tid; i < 64 * D_; i += 128)
    sA[0][i] = (__bf16)x[(size_t)t0 * D_ + i];
  {
    const uint4* s4 = (const uint4*)wbt;
    uint4* d4 = (uint4*)&sB[0][0];
#pragma unroll
    for (int i = 0; i < 16; ++i) stage_b128(s4 + tid + i * 128, d4 + tid + i * 128);
  }
  stage_wait();
  __syncthreads();
  v8f acc[4][2];
#pragma unroll
  for (int mt = 0; mt < 4; ++mt)
#pragma unroll
    for (int nt = 0; nt < 2; ++nt) acc[mt][nt] = vzero();
#pragma unroll
  for (int ks = 0; ks < 4; ++ks) {
    v16bf bf[2];
#pragma unroll
    for (int nt = 0; nt < 2; ++nt)
      bf[nt] = load_bt_frag(&sB[wave * 32 + nt * 16][ks * 32], D_);
#pragma unroll
    for (int mt = 0; mt < 4; ++mt) {
      v16bf a = load_a_frag(&sA[mt * 16][ks * 32], D_);
#pragma unroll
      for (int nt = 0; nt < 2; ++nt)
        acc[mt][nt] = wmma_bf16(a, bf[nt], acc[mt][nt]);
    }
  }
  int n = lane & 15, mh = 8 * (lane >> 4);
#pragma unroll
  for (int mt = 0; mt < 4; ++mt)
#pragma unroll
    for (int nt = 0; nt < 2; ++nt) {
      int col = wave * 32 + nt * 16 + n;
      float bb = bias[col];
#pragma unroll
      for (int r = 0; r < 8; ++r)
        h[(size_t)(t0 + mt * 16 + mh + r) * D_ + col] = acc[mt][nt][r] + bb;
    }
}

// ------------------------- spectral + m_phi + AR ---------------------------
// delta[b, t0..t0+63, :] = sum_k (Lk @ hn_b) @ m_phi_k + sum_tap shift @ mu_tap
// Stage 1: A frags from precomputed per-diagonal table (global, L2-resident),
//          B frags from hnT (global, L2-resident). No LDS, no barriers.
// Stage 2: A from LDS sG (conv result), B frags direct from mphi_bt/mu_bt.

__global__ void __launch_bounds__(128)
k_spec(const __bf16* __restrict__ hnAll, const __bf16* __restrict__ hnTAll,
       const uint4* __restrict__ atab, const __bf16* __restrict__ mphi,
       const __bf16* __restrict__ muB, float* __restrict__ delta) {
  __shared__ __attribute__((aligned(16))) __bf16 sG[64][D_];   // 16 KB
  int b  = blockIdx.y;
  int t0 = blockIdx.x * 64;
  const __bf16* hn  = hnAll + (size_t)b * T_ * D_;
  const __bf16* hnT = hnTAll + (size_t)b * T_ * D_;
  int tid = threadIdx.x, wave = tid >> 5, lane = tid & 31;
  int sEnd = t0 + 64;

  v8f dacc[4][2];
#pragma unroll
  for (int mt = 0; mt < 4; ++mt)
#pragma unroll
    for (int nt = 0; nt < 2; ++nt) dacc[mt][nt] = vzero();

  for (int kk = 0; kk < KE + KU; ++kk) {
    if (kk < KE) {
      v8f gacc[4][2];
#pragma unroll
      for (int mt = 0; mt < 4; ++mt)
#pragma unroll
        for (int nt = 0; nt < 2; ++nt) gacc[mt][nt] = vzero();

      for (int s0 = 0; s0 < sEnd; s0 += 32) {
        int dg = ((t0 - s0) >> 5) + 1;   // 0..63
        v16bf b0 = load_bt_frag(hnT + (size_t)(wave * 32) * T_ + s0, T_);
        v16bf b1 = load_bt_frag(hnT + (size_t)(wave * 32 + 16) * T_ + s0, T_);
#pragma unroll
        for (int mt = 0; mt < 4; ++mt) {
          size_t fi = (((size_t)(kk * NDG + dg) * 4 + mt) * 32 + lane) * 2;
          AFrag f;
          f.q[0] = atab[fi];
          f.q[1] = atab[fi + 1];
          gacc[mt][0] = wmma_bf16(f.v, b0, gacc[mt][0]);
          gacc[mt][1] = wmma_bf16(f.v, b1, gacc[mt][1]);
        }
      }
      __syncthreads();  // previous stage-2 readers of sG done
      {
        int n = lane & 15, mh = 8 * (lane >> 4);
#pragma unroll
        for (int mt = 0; mt < 4; ++mt)
#pragma unroll
          for (int nt = 0; nt < 2; ++nt) {
            int col = wave * 32 + nt * 16 + n;
#pragma unroll
            for (int r = 0; r < 8; ++r)
              sG[mt * 16 + mh + r][col] = (__bf16)gacc[mt][nt][r];
          }
      }
    } else {
      // AR tap: "G" = hn rows shifted by tap (zeros below t=0)
      int tap = kk - KE;
      __syncthreads();
      unsigned* gdst = (unsigned*)&sG[0][0];
      for (int i = tid; i < 4096; i += 128) {
        int row = i >> 6, cu = i & 63;
        int ts = t0 + row - tap;
        gdst[i] = (ts >= 0) ? ((const unsigned*)(hn + (size_t)ts * D_))[cu] : 0u;
      }
    }
    __syncthreads();  // sG ready
    const __bf16* wsrc = (kk < KE) ? (mphi + (size_t)kk * D_ * D_)
                                   : (muB + (size_t)(kk - KE) * D_ * D_);
    if (kk + 1 < KE + KU) {
      const __bf16* nxt = (kk + 1 < KE) ? (mphi + (size_t)(kk + 1) * D_ * D_)
                                        : (muB + (size_t)(kk + 1 - KE) * D_ * D_);
      __builtin_prefetch(nxt, 0, 3);
    }
    // stage 2: dacc += G[64x128] @ W[128x128], B frags straight from L2
#pragma unroll
    for (int ks = 0; ks < 4; ++ks) {
      v16bf bf0 = load_bt_frag(wsrc + (size_t)(wave * 32) * D_ + ks * 32, D_);
      v16bf bf1 = load_bt_frag(wsrc + (size_t)(wave * 32 + 16) * D_ + ks * 32, D_);
#pragma unroll
      for (int mt = 0; mt < 4; ++mt) {
        v16bf a = load_a_frag(&sG[mt * 16][ks * 32], D_);
        dacc[mt][0] = wmma_bf16(a, bf0, dacc[mt][0]);
        dacc[mt][1] = wmma_bf16(a, bf1, dacc[mt][1]);
      }
    }
  }
  int n = lane & 15, mh = 8 * (lane >> 4);
  float* dl = delta + (size_t)b * T_ * D_;
#pragma unroll
  for (int mt = 0; mt < 4; ++mt)
#pragma unroll
    for (int nt = 0; nt < 2; ++nt) {
      int col = wave * 32 + nt * 16 + n;
#pragma unroll
      for (int r = 0; r < 8; ++r)
        dl[(size_t)(t0 + mt * 16 + mh + r) * D_ + col] = dacc[mt][nt][r];
    }
}

// ------------------------------ batched scan -------------------------------

__global__ void __launch_bounds__(256)
k_scan(const float* __restrict__ delta, const __bf16* __restrict__ wscanT,
       float* __restrict__ y) {
  __shared__ __attribute__((aligned(16))) __bf16 zb[2][16][D_];  // 8 KB
  int tid = threadIdx.x, wave = tid >> 5, lane = tid & 31;
  int n = lane & 15, mh = 8 * (lane >> 4);

  v16bf wf[8];
#pragma unroll
  for (int ks = 0; ks < 8; ++ks)
    wf[ks] = load_bt_frag(wscanT + (size_t)(wave * 16) * 256 + ks * 32, 256);

  for (int i = tid; i < 2 * 16 * D_; i += 256) ((__bf16*)zb)[i] = (__bf16)0.0f;
  __syncthreads();

  int cur = 0;
  for (int t = 0; t < T_; ++t) {
    v8f acc;
#pragma unroll
    for (int r = 0; r < 8; ++r) {
      int m = mh + r;
      acc[r] = (m < 8) ? delta[((size_t)m * T_ + t) * D_ + wave * 16 + n] : 0.0f;
    }
#pragma unroll
    for (int ks = 0; ks < 8; ++ks) {
      const __bf16* base =
          (ks < 4) ? &zb[cur][0][ks * 32] : &zb[cur ^ 1][0][(ks - 4) * 32];
      v16bf a = load_a_frag(base, D_);
      acc = wmma_bf16(a, wf[ks], acc);
    }
    __syncthreads();
#pragma unroll
    for (int r = 0; r < 8; ++r) {
      int m = mh + r;
      if (m < 8) {
        y[((size_t)m * T_ + t) * D_ + wave * 16 + n] = acc[r];
        zb[cur ^ 1][m][wave * 16 + n] = (__bf16)acc[r];
      }
    }
    __syncthreads();
    cur ^= 1;
  }
}

// ------------------------- gelu -> w1 -> GLU + residual --------------------

__global__ void __launch_bounds__(128)
k_w1(const float* __restrict__ yin, const __bf16* __restrict__ w1bt,
     const float* __restrict__ b1, float* __restrict__ h) {
  __shared__ __attribute__((aligned(16))) __bf16 sA[64][D_];     // 16 KB
  __shared__ __attribute__((aligned(16))) __bf16 sB[256][32];    // 16 KB
  int t0 = blockIdx.x * 64;
  int tid = threadIdx.x, wave = tid >> 5, lane = tid & 31;
  for (int i = tid; i < 64 * D_; i += 128) {
    float v = yin[(size_t)t0 * D_ + i];
    sA[0][i] = (__bf16)(0.5f * v * (1.0f + erff(v * 0.70710678118f)));
  }
  v8f acc[4][4];
#pragma unroll
  for (int mt = 0; mt < 4; ++mt)
#pragma unroll
    for (int j = 0; j < 4; ++j) acc[mt][j] = vzero();

  for (int ks = 0; ks < 4; ++ks) {
    __syncthreads();
    {
      uint4* d4 = (uint4*)&sB[0][0];
#pragma unroll
      for (int i = 0; i < 8; ++i) {
        int idx = tid + i * 128;
        int nrow = idx >> 2, c = idx & 3;
        stage_b128(((const uint4*)(w1bt + (size_t)nrow * D_ + ks * 32)) + c,
                   d4 + idx);
      }
    }
    stage_wait();
    __syncthreads();
    v16bf bf[4];
#pragma unroll
    for (int j = 0; j < 4; ++j) {
      int nt = (j < 2) ? (2 * wave + j) : (2 * wave + j + 6);
      bf[j] = load_bt_frag(&sB[nt * 16][0], 32);
    }
#pragma unroll
    for (int mt = 0; mt < 4; ++mt) {
      v16bf a = load_a_frag(&sA[mt * 16][ks * 32], D_);
#pragma unroll
      for (int j = 0; j < 4; ++j)
        acc[mt][j] = wmma_bf16(a, bf[j], acc[mt][j]);
    }
  }
  int n = lane & 15, mh = 8 * (lane >> 4);
#pragma unroll
  for (int mt = 0; mt < 4; ++mt)
#pragma unroll
    for (int p = 0; p < 2; ++p) {
      int cola = (2 * wave + p) * 16 + n;
      int colg = cola + D_;
      float ba = b1[cola], bg = b1[colg];
#pragma unroll
      for (int r = 0; r < 8; ++r) {
        size_t idx = (size_t)(t0 + mt * 16 + mh + r) * D_ + cola;
        float a = acc[mt][p][r] + ba;
        float g = acc[mt][p + 2][r] + bg;
        h[idx] = a * (1.0f / (1.0f + expf(-g))) + h[idx];
      }
    }
}

// ------------------------------ output proj --------------------------------

__global__ void __launch_bounds__(128)
k_proj(const float* __restrict__ h, const __bf16* __restrict__ wbt,
       const float* __restrict__ bias, float* __restrict__ out) {
  __shared__ __attribute__((aligned(16))) __bf16 sA[64][D_];   // 16 KB
  __shared__ __attribute__((aligned(16))) __bf16 sB[DT][D_];   // 16 KB
  int t0 = blockIdx.x * 64;
  int tid = threadIdx.x, wave = tid >> 5, lane = tid & 31;
  for (int i = tid; i < 64 * D_; i += 128)
    sA[0][i] = (__bf16)h[(size_t)t0 * D_ + i];
  {
    const uint4* s4 = (const uint4*)wbt;
    uint4* d4 = (uint4*)&sB[0][0];
#pragma unroll
    for (int i = 0; i < 8; ++i) stage_b128(s4 + tid + i * 128, d4 + tid + i * 128);
  }
  stage_wait();
  __syncthreads();
  v8f acc[4];
#pragma unroll
  for (int mt = 0; mt < 4; ++mt) acc[mt] = vzero();
#pragma unroll
  for (int ks = 0; ks < 4; ++ks) {
    v16bf bf = load_bt_frag(&sB[wave * 16][ks * 32], D_);
#pragma unroll
    for (int mt = 0; mt < 4; ++mt) {
      v16bf a = load_a_frag(&sA[mt * 16][ks * 32], D_);
      acc[mt] = wmma_bf16(a, bf, acc[mt]);
    }
  }
  int n = lane & 15, mh = 8 * (lane >> 4);
  int col = wave * 16 + n;
  float bb = bias[col];
#pragma unroll
  for (int mt = 0; mt < 4; ++mt)
#pragma unroll
    for (int r = 0; r < 8; ++r)
      out[(size_t)(t0 + mt * 16 + mh + r) * DT + col] = acc[mt][r] + bb;
}

// ------------------------------ launcher -----------------------------------

extern "C" void kernel_launch(void* const* d_in, const int* in_sizes, int n_in,
                              void* d_out, int out_size, void* d_ws, size_t ws_size,
                              hipStream_t stream) {
  (void)in_sizes; (void)n_in; (void)out_size; (void)ws_size;
  const float* x      = (const float*)d_in[0];
  const float* emb_w  = (const float*)d_in[1];
  const float* emb_b  = (const float*)d_in[2];
  const float* ln_g   = (const float*)d_in[3];
  const float* ln_b   = (const float*)d_in[4];
  const float* m_phi  = (const float*)d_in[5];
  const float* m_u    = (const float*)d_in[6];
  const float* m_y    = (const float*)d_in[7];
  const float* w1     = (const float*)d_in[8];
  const float* b1     = (const float*)d_in[9];
  const float* proj_w = (const float*)d_in[10];
  const float* proj_b = (const float*)d_in[11];
  const float* evals  = (const float*)d_in[12];
  const float* evecs  = (const float*)d_in[13];

  char* p = (char*)d_ws;
  auto alloc = [&](size_t bytes) {
    char* r = p;
    p += (bytes + 255) & ~(size_t)255;
    return r;
  };
  float*    h       = (float*)alloc((size_t)ROWS * D_ * 4);
  float*    yb      = (float*)alloc((size_t)ROWS * D_ * 4);
  float*    delta   = (float*)alloc((size_t)ROWS * D_ * 4);
  __bf16*   hn      = (__bf16*)alloc((size_t)ROWS * D_ * 2);
  __bf16*   hnT     = (__bf16*)alloc((size_t)ROWS * D_ * 2);
  __bf16*   embw_bt = (__bf16*)alloc((size_t)D_ * D_ * 2);
  __bf16*   mphi_bt = (__bf16*)alloc((size_t)LL * KE * D_ * D_ * 2);
  __bf16*   mu_bt   = (__bf16*)alloc((size_t)LL * KU * D_ * D_ * 2);
  __bf16*   wsc_bt  = (__bf16*)alloc((size_t)LL * D_ * 2 * D_ * 2);
  __bf16*   w1_bt   = (__bf16*)alloc((size_t)LL * D_ * 2 * D_ * 2);
  __bf16*   pw_bt   = (__bf16*)alloc((size_t)D_ * DT * 2);
  __bf16*   eig_bf  = (__bf16*)alloc((size_t)KE * T_ * 2);
  unsigned* atab    = (unsigned*)alloc((size_t)KE * NDG * 4 * 32 * 8 * 4);

  k_prep_embw<<<64, 256, 0, stream>>>(emb_w, embw_bt);
  k_prep_mphi<<<768, 256, 0, stream>>>(m_phi, mphi_bt);
  k_prep_mu<<<384, 256, 0, stream>>>(m_u, mu_bt);
  k_cvt<<<512, 256, 0, stream>>>(m_y, wsc_bt, LL * D_ * 2 * D_);  // native layout
  k_prep_w1<<<512, 256, 0, stream>>>(w1, w1_bt);
  k_prep_proj<<<32, 256, 0, stream>>>(proj_w, pw_bt);
  k_prep_eig<<<192, 256, 0, stream>>>(evals, evecs, eig_bf);
  k_prep_afrag<<<3072, 256, 0, stream>>>(eig_bf, atab);

  k_emb<<<ROWS / 64, 128, 0, stream>>>(x, embw_bt, emb_b, h);

  for (int l = 0; l < LL; ++l) {
    k_ln<<<ROWS / 8, 256, 0, stream>>>(h, ln_g + l * D_, ln_b + l * D_, hn);
    k_tr<<<dim3(T_ / 64, B_), 128, 0, stream>>>(hn, hnT);
    k_spec<<<dim3(T_ / 64, B_), 128, 0, stream>>>(
        hn, hnT, (const uint4*)atab, mphi_bt + (size_t)l * KE * D_ * D_,
        mu_bt + (size_t)l * KU * D_ * D_, delta);
    k_scan<<<1, 256, 0, stream>>>(delta, wsc_bt + (size_t)l * D_ * 2 * D_, yb);
    k_w1<<<ROWS / 64, 128, 0, stream>>>(yb, w1_bt + (size_t)l * D_ * 2 * D_,
                                        b1 + l * 2 * D_, h);
  }
  k_proj<<<ROWS / 64, 128, 0, stream>>>(h, pw_bt, proj_b, (float*)d_out);
}
